// CausalSelfAttention_76184129896687
// MI455X (gfx1250) — matrix operand order repaired and checked by reading, not compiled, not run on hardware
//
#include <hip/hip_runtime.h>

// ---------------------------------------------------------------------------
// CDNA5 (gfx1250) causal self-attention: bf16 WMMA everywhere.
//   kernel 1: qkv = x @ W_qkv^T  -> scatter into head-major bf16 Q,K,V
//   kernel 2: flash attention (causal, online softmax), WMMA for QK^T and PV,
//             K chunk staged via GLOBAL_LOAD_ASYNC_TO_LDS_B128 (ASYNCcnt)
//   kernel 3: out = y @ W_proj^T (f32 out)
// ---------------------------------------------------------------------------

typedef __attribute__((ext_vector_type(16))) __bf16 v16bf;
typedef __attribute__((ext_vector_type(8)))  __bf16 v8bf;
typedef __attribute__((ext_vector_type(4)))  __bf16 v4bf;
typedef __attribute__((ext_vector_type(8)))  float  v8f;

union Frag16 { v16bf v; v8bf h[2]; };

__device__ __forceinline__ __bf16 f2bf(float f) {
  union { float f; unsigned u; } c; c.f = f;
  unsigned r = c.u + 0x7FFFu + ((c.u >> 16) & 1u);   // round-to-nearest-even
  unsigned short s = (unsigned short)(r >> 16);
  return __builtin_bit_cast(__bf16, s);
}

// Two contiguous 16B runs -> one 16-element bf16 fragment (matches CDNA5
// 16-bit A/B VGPR layouts when p0/p1 are chosen per lane-half).
__device__ __forceinline__ v16bf ld_frag(const __bf16* p0, const __bf16* p1) {
  Frag16 f;
  f.h[0] = *(const v8bf*)p0;
  f.h[1] = *(const v8bf*)p1;
  return f.v;
}

__device__ __forceinline__ v8f wmma_bf16(v16bf a, v16bf b, v8f c) {
  return __builtin_amdgcn_wmma_f32_16x16x32_bf16(
      /*neg_a=*/false, a, /*neg_b=*/false, b,
      /*c_mod=*/(short)0, c, /*reuse_a=*/false, /*reuse_b=*/false);
}

// Async global->LDS copy of 16 bytes per lane (tracked on ASYNCcnt).
__device__ __forceinline__ void async_copy_b128(unsigned lds_byte_addr,
                                                unsigned glb_byte_off,
                                                const void* sbase) {
  asm volatile("global_load_async_to_lds_b128 %0, %1, %2"
               :
               : "v"(lds_byte_addr), "v"(glb_byte_off),
                 "s"((unsigned long long)sbase)
               : "memory");
}
__device__ __forceinline__ void wait_async0() {
  asm volatile("s_wait_asynccnt 0x0" ::: "memory");
}

// ---------------------------------------------------------------------------
// GEMM: C[m,n] = sum_k A[m,k] * W[n,k]   (A row-major [M,K], W row-major [N,K])
// Block tile 128x128x64, 256 threads (8 waves, 2x4), wave tile 64x32.
// MODE 0: scatter into q/k/v [B,H,T,64] bf16 (T=2048,H=16,D=1024), q scaled.
// MODE 1: plain f32 output row-major [M,N].
// ABF: A is bf16 (true) or f32 (false).
// ---------------------------------------------------------------------------
template <int MODE, bool ABF>
__global__ __launch_bounds__(256) void gemm_bf16_wmma(
    const void* __restrict__ Av, const float* __restrict__ W,
    float* __restrict__ Cout,
    __bf16* __restrict__ qo, __bf16* __restrict__ ko, __bf16* __restrict__ vo,
    int M, int N, int K) {
  constexpr int LDT = 72;                         // padded bf16 tile stride
  __shared__ __align__(16) __bf16 As[128 * LDT];  // 128 x 64
  __shared__ __align__(16) __bf16 Bs[128 * LDT];

  const int tid  = threadIdx.x;
  const int lane = tid & 31;
  const int wv   = tid >> 5;                  // wave 0..7
  const int wm   = wv >> 2;                   // 0..1 (M)
  const int wn   = wv & 3;                    // 0..3 (N)
  const int hl   = (lane >> 4) & 1;           // lane half
  const int l15  = lane & 15;
  const int bm   = blockIdx.y * 128;
  const int bn   = blockIdx.x * 128;

  v8f acc[4][2] = {};

  for (int kt = 0; kt < K; kt += 64) {
    // ---- stage A tile (128x64) -> bf16 LDS ----
    if (ABF) {
      const __bf16* Ab = (const __bf16*)Av;
#pragma unroll
      for (int i = 0; i < 4; ++i) {
        int idx = tid + i * 256;                 // 0..1023, 8x v8bf per row
        int row = idx >> 3, c8 = (idx & 7) << 3;
        *(v8bf*)&As[row * LDT + c8] =
            *(const v8bf*)(Ab + (long)(bm + row) * K + kt + c8);
      }
      if (kt + 64 < K)
        __builtin_prefetch(Ab + (long)(bm + (tid >> 1)) * K + kt + 64, 0, 1);
    } else {
      const float* Af = (const float*)Av;
#pragma unroll
      for (int i = 0; i < 8; ++i) {
        int idx = tid + i * 256;                 // 0..2047, 16x float4 per row
        int row = idx >> 4, c4 = (idx & 15) << 2;
        const float4 fa = *(const float4*)(Af + (long)(bm + row) * K + kt + c4);
        v4bf t;
        t[0] = f2bf(fa.x); t[1] = f2bf(fa.y); t[2] = f2bf(fa.z); t[3] = f2bf(fa.w);
        *(v4bf*)&As[row * LDT + c4] = t;
      }
      if (kt + 64 < K)
        __builtin_prefetch(Af + (long)(bm + (tid >> 1)) * K + kt + 64, 0, 1);
    }
    // ---- stage B tile (128 output cols x 64) -> bf16 LDS ----
#pragma unroll
    for (int i = 0; i < 8; ++i) {
      int idx = tid + i * 256;
      int row = idx >> 4, c4 = (idx & 15) << 2;
      const float4 fb = *(const float4*)(W + (long)(bn + row) * K + kt + c4);
      v4bf t;
      t[0] = f2bf(fb.x); t[1] = f2bf(fb.y); t[2] = f2bf(fb.z); t[3] = f2bf(fb.w);
      *(v4bf*)&Bs[row * LDT + c4] = t;
    }
    if (kt + 64 < K)
      __builtin_prefetch(W + (long)(bn + (tid >> 1)) * K + kt + 64, 0, 1);
    __syncthreads();

    // ---- two 32-wide K-steps, 8 WMMAs each ----
#pragma unroll
    for (int kk = 0; kk < 2; ++kk) {
      v16bf af[4], bfr[2];
#pragma unroll
      for (int mi = 0; mi < 4; ++mi) {
        const __bf16* p = &As[(wm * 64 + mi * 16 + l15) * LDT + kk * 32];
        af[mi] = ld_frag(p + hl * 8, p + 16 + hl * 8);        // A layout
      }
#pragma unroll
      for (int ni = 0; ni < 2; ++ni) {
        const __bf16* p = &Bs[(wn * 32 + ni * 16 + l15) * LDT + kk * 32];
        bfr[ni] = ld_frag(p + hl * 8, p + 16 + hl * 8);       // B layout
      }
#pragma unroll
      for (int mi = 0; mi < 4; ++mi)
#pragma unroll
        for (int ni = 0; ni < 2; ++ni)
          acc[mi][ni] = wmma_bf16(af[mi], bfr[ni], acc[mi][ni]);
    }
    __syncthreads();
  }

  // ---- epilogue ----
#pragma unroll
  for (int mi = 0; mi < 4; ++mi)
#pragma unroll
    for (int ni = 0; ni < 2; ++ni)
#pragma unroll
      for (int r = 0; r < 8; ++r) {
        int m = bm + wm * 64 + mi * 16 + r + hl * 8;
        int n = bn + wn * 32 + ni * 16 + l15;
        float val = acc[mi][ni][r];
        if constexpr (MODE == 0) {
          int b = m >> 11, t = m & 2047;       // T = 2048
          int part = n >> 10, cc = n & 1023;   // D = 1024
          int hh = cc >> 6, hd = cc & 63;      // HD = 64
          long off = (((long)(b * 16 + hh)) * 2048 + t) * 64 + hd;
          if (part == 0)      qo[off] = f2bf(val * 0.125f);  // fold 1/sqrt(64)
          else if (part == 1) ko[off] = f2bf(val);
          else                vo[off] = f2bf(val);
        } else {
          Cout[(long)m * N + n] = val;
        }
      }
}

// ---------------------------------------------------------------------------
// Flash attention: one block per (b, h, 64-query tile); 4 waves x 16 q-rows.
// S = Q K^T and O += P V via bf16 WMMA; online softmax with shfl_xor row
// reductions (a C-layout row lives in one 16-lane half of the wave32).
// K chunk: async global->LDS copy. V chunk: load + register transpose.
// ---------------------------------------------------------------------------
__global__ __launch_bounds__(128) void attn_fwd(
    const __bf16* __restrict__ qg, const __bf16* __restrict__ kg,
    const __bf16* __restrict__ vg, __bf16* __restrict__ yo) {
  const int T = 2048, HD = 64, H = 16, D = 1024;
  const int qt = blockIdx.x;
  const int hh = blockIdx.y;
  const int bb = blockIdx.z;
  const int bh = bb * H + hh;
  const int tid = threadIdx.x;
  const int wv = tid >> 5;
  const int lane = tid & 31;
  const int hl = lane >> 4, l15 = lane & 15;
  const int qbase = qt * 64;
  const int qrow = qbase + wv * 16;

  __shared__ __align__(16) __bf16 Kt[64 * 72];       // [key][hd]
  __shared__ __align__(16) __bf16 Vt[64 * 72];       // [hd][key] (transposed)
  __shared__ __align__(16) __bf16 Pw[4 * 16 * 72];   // wave-private P staging

  // Q fragments for this wave's 16 rows (resident all kernel).
  const __bf16* qptr = qg + ((long)bh * T + qrow + l15) * HD;
  v16bf qf0 = ld_frag(qptr + hl * 8, qptr + 16 + hl * 8);        // hd 0..31
  v16bf qf1 = ld_frag(qptr + 32 + hl * 8, qptr + 48 + hl * 8);   // hd 32..63

  float mrow[8], lrow[8];
  v8f oacc[4] = {};
#pragma unroll
  for (int r = 0; r < 8; ++r) { mrow[r] = -1e30f; lrow[r] = 0.0f; }

  __bf16* pw = &Pw[wv * 16 * 72];

  for (int kc = 0; kc <= qt; ++kc) {
    const int kbase = kc * 64;
    const __bf16* kcp = kg + ((long)bh * T + kbase) * HD;
    const __bf16* vcp = vg + ((long)bh * T + kbase) * HD;
#pragma unroll
    for (int i = 0; i < 4; ++i) {
      int idx = tid * 4 + i;                 // 0..511 over 64x(8x8) chunks
      int row = idx >> 3, c8 = (idx & 7) << 3;
      // K: straight async copy into LDS, no layout change (ASYNCcnt).
      async_copy_b128((unsigned)(uintptr_t)&Kt[row * 72 + c8],
                      (unsigned)((row * HD + c8) * 2), kcp);
      // V: needs [hd][key] transpose -> regular load + bf16 scatter.
      v8bf tv = *(const v8bf*)(vcp + row * HD + c8);
#pragma unroll
      for (int j = 0; j < 8; ++j) Vt[(c8 + j) * 72 + row] = tv[j];
    }
    wait_async0();
    __syncthreads();

    // ---- S = Q K^T (4 key subtiles x 2 k-steps = 8 WMMAs) ----
    v8f sacc[4] = {};
#pragma unroll
    for (int s = 0; s < 4; ++s) {
      const __bf16* p = &Kt[(s * 16 + l15) * 72];
      v16bf kf0 = ld_frag(p + hl * 8, p + 16 + hl * 8);
      v16bf kf1 = ld_frag(p + 32 + hl * 8, p + 48 + hl * 8);
      sacc[s] = wmma_bf16(qf0, kf0, sacc[s]);
      sacc[s] = wmma_bf16(qf1, kf1, sacc[s]);
    }

    // ---- causal mask on the diagonal chunk ----
    if (kc == qt) {
#pragma unroll
      for (int s = 0; s < 4; ++s)
#pragma unroll
        for (int r = 0; r < 8; ++r) {
          int key = kbase + s * 16 + l15;
          int qi = qrow + r + hl * 8;
          if (key > qi) sacc[s][r] = -1e30f;
        }
    }

    // ---- online softmax (row reductions across 16 lanes of each half) ----
    float pr[4][8];
#pragma unroll
    for (int r = 0; r < 8; ++r) {
      float mx = fmaxf(fmaxf(sacc[0][r], sacc[1][r]),
                       fmaxf(sacc[2][r], sacc[3][r]));
#pragma unroll
      for (int msk = 8; msk >= 1; msk >>= 1)
        mx = fmaxf(mx, __shfl_xor(mx, msk, 32));
      float mnew = fmaxf(mrow[r], mx);
      float corr = __expf(mrow[r] - mnew);
      float rs = 0.0f;
#pragma unroll
      for (int s = 0; s < 4; ++s) {
        float p = __expf(sacc[s][r] - mnew);
        pr[s][r] = p;
        rs += p;
      }
#pragma unroll
      for (int msk = 8; msk >= 1; msk >>= 1) rs += __shfl_xor(rs, msk, 32);
      lrow[r] = lrow[r] * corr + rs;
      mrow[r] = mnew;
#pragma unroll
      for (int ni = 0; ni < 4; ++ni) oacc[ni][r] *= corr;
    }

    // ---- P: C-layout -> A-layout via wave-private LDS ----
#pragma unroll
    for (int s = 0; s < 4; ++s)
#pragma unroll
      for (int r = 0; r < 8; ++r)
        pw[(r + hl * 8) * 72 + s * 16 + l15] = f2bf(pr[s][r]);

    const __bf16* pp = pw + l15 * 72;
    v16bf pf0 = ld_frag(pp + hl * 8, pp + 16 + hl * 8);        // keys 0..31
    v16bf pf1 = ld_frag(pp + 32 + hl * 8, pp + 48 + hl * 8);   // keys 32..63

    // ---- O += P V (4 hd subtiles x 2 k-steps = 8 WMMAs) ----
#pragma unroll
    for (int ni = 0; ni < 4; ++ni) {
      const __bf16* p = &Vt[(ni * 16 + l15) * 72];
      v16bf vf0 = ld_frag(p + hl * 8, p + 16 + hl * 8);
      v16bf vf1 = ld_frag(p + 32 + hl * 8, p + 48 + hl * 8);
      oacc[ni] = wmma_bf16(pf0, vf0, oacc[ni]);
      oacc[ni] = wmma_bf16(pf1, vf1, oacc[ni]);
    }
    __syncthreads();
  }

  // ---- normalize + write y (bf16, [B,T,D] with head hh at cols hh*64..) ----
#pragma unroll
  for (int ni = 0; ni < 4; ++ni)
#pragma unroll
    for (int r = 0; r < 8; ++r) {
      int row = qrow + r + hl * 8;
      float val = oacc[ni][r] / lrow[r];
      yo[((long)bb * T + row) * D + hh * HD + ni * 16 + l15] = f2bf(val);
    }
}

// ---------------------------------------------------------------------------
extern "C" void kernel_launch(void* const* d_in, const int* in_sizes, int n_in,
                              void* d_out, int out_size, void* d_ws,
                              size_t ws_size, hipStream_t stream) {
  const float* x      = (const float*)d_in[0];
  const float* w_qkv  = (const float*)d_in[1];
  const float* w_proj = (const float*)d_in[2];
  float* out = (float*)d_out;

  const int Bsz = 2, T = 2048, D = 1024;
  const long nEl = (long)Bsz * T * D;          // 4M elements per buffer

  __bf16* qb = (__bf16*)d_ws;                  // [B,H,T,64] bf16, 8MB
  __bf16* kb = qb + nEl;
  __bf16* vb = kb + nEl;
  __bf16* yb = vb + nEl;                       // [B,T,D] bf16

  // 1) QKV projection + scatter to head-major Q/K/V (Q pre-scaled by 1/8)
  dim3 g1(3 * D / 128, Bsz * T / 128);
  gemm_bf16_wmma<0, false><<<g1, dim3(256), 0, stream>>>(
      (const void*)x, w_qkv, nullptr, qb, kb, vb, Bsz * T, 3 * D, D);

  // 2) Causal flash attention
  dim3 g2(T / 64, 16, Bsz);
  attn_fwd<<<g2, dim3(128), 0, stream>>>(qb, kb, vb, yb);

  // 3) Output projection -> f32 d_out
  dim3 g3(D / 128, Bsz * T / 128);
  gemm_bf16_wmma<1, true><<<g3, dim3(256), 0, stream>>>(
      (const void*)yb, w_proj, out, nullptr, nullptr, nullptr, Bsz * T, D, D);
}
